// ResidualVQ_2D_79963701117575
// MI455X (gfx1250) — compile-verified, matrix operand order checked by reading, not run on hardware
//
#include <hip/hip_runtime.h>
#include <hip/hip_bf16.h>
#include <math.h>

// ---------------- problem constants ----------------
#define BDIM   32
#define CDIM   128
#define JDIM   24
#define TDIM   128
#define QSTAGES 6
#define KCODES 1024
#define MROWS  (BDIM * JDIM * TDIM)      // 98304 tokens
#define MTILES (MROWS / 16)              // 6144 16-row tiles
#define NQOUT  ((size_t)BDIM * CDIM * JDIM * TDIM)    // 12,582,912
#define NIDX   ((size_t)BDIM * JDIM * TDIM * QSTAGES) // 589,824
#define EPSV   1e-10f
#define TILE_USH 2048                    // ushorts per 16-code B tile: 4*32*16

// ---------------- vector types ----------------
typedef __attribute__((ext_vector_type(16))) __bf16        v16bf;
typedef __attribute__((ext_vector_type(8)))  float         v8f;
typedef __attribute__((ext_vector_type(4)))  unsigned int  v4u;

union BF16x16 {           // 32 bytes: two b128 loads <-> one WMMA operand
    v4u            u[2];
    unsigned short s[16];
    v16bf          v;
};

// float -> bf16, round-to-nearest-even
__device__ __forceinline__ unsigned short f2bf(float f) {
    unsigned int u = __float_as_uint(f);
    u += 0x7FFFu + ((u >> 16) & 1u);
    return (unsigned short)(u >> 16);
}

// A-matrix (16x32 bf16) lane layout per CDNA5 ISA 7.12.2:
// lanes 0-15 hold K={0..7,16..23}, lanes 16-31 hold K={8..15,24..31}
__device__ __forceinline__ size_t apack_off(int m, int c) {
    int tile = m >> 4, Mi = m & 15;
    int cc = c >> 5, kk = c & 31;
    int half = (kk >> 3) & 1;
    int lane = Mi + (half << 4);
    int elem = (kk & 7) | ((kk >> 4) << 3);
    return ((size_t)(tile * 4 + cc) * 32 + lane) * 16 + elem;
}
// B tile lane layout (mirror of A with N across lanes)
__device__ __forceinline__ int bpack_elem_off(int n, int kk) {
    int half = (kk >> 3) & 1;
    int lane = n + (half << 4);
    int elem = (kk & 7) | ((kk >> 4) << 3);
    return lane * 16 + elem;
}

// ---------------- packing kernels ----------------
__global__ __launch_bounds__(256) void rvq_pack_x(const float* __restrict__ x,
                                                  float* __restrict__ res,
                                                  unsigned short* __restrict__ apack) {
    int gid = blockIdx.x * 256 + threadIdx.x;         // over M*C
    int m = gid / CDIM, c = gid - m * CDIM;
    int b = m / (JDIM * TDIM);
    int rjt = m - b * (JDIM * TDIM);
    int j = rjt / TDIM, t = rjt - j * TDIM;
    float v = x[(((size_t)b * CDIM + c) * JDIM + j) * TDIM + t];
    res[(size_t)m * CDIM + c] = v;
    apack[apack_off(m, c)] = f2bf(v);
}

__global__ __launch_bounds__(256) void rvq_pack_cb(const float* __restrict__ cb,
                                                   unsigned short* __restrict__ bpack,
                                                   float* __restrict__ cnorm) {
    int gid = blockIdx.x * 256 + threadIdx.x;         // q*K + k
    if (gid >= QSTAGES * KCODES) return;
    int q = gid / KCODES, k = gid - q * KCODES;
    int kb = k >> 4, n = k & 15;
    const float* row = cb + (size_t)gid * CDIM;
    unsigned short* bt = bpack + (size_t)(q * 64 + kb) * TILE_USH;
    float s = 0.f;
    for (int c = 0; c < CDIM; ++c) {
        float v = row[c];
        s += v * v;
        int cc = c >> 5, kk = c & 31;
        bt[cc * (32 * 16) + bpack_elem_off(n, kk)] = f2bf(v);
    }
    cnorm[gid] = s;
}

// ---------------- stage kernels ----------------
// Each wave: one 16-row M tile vs all K=1024 codes via bf16 WMMA.
// acc init = -0.5*||c_k||^2 ; argmax(r.c - 0.5||c||^2) == argmin distance.
// Double-buffered LDS B tiles, register-resident A, 4 back-to-back WMMAs.
__global__ __launch_bounds__(256) void rvq_dist_argmin(
        const unsigned short* __restrict__ apack,   // [Mtile][4][32][16] bf16
        const unsigned short* __restrict__ bpack,   // [Q][64][4][32][16] bf16
        const float* __restrict__ cnorm,            // [Q][K]
        int* __restrict__ idx_all,                  // [M][Q]
        int q) {
    __shared__ v4u shB[2][256];                     // two 16-code B tiles (8KB)
    __shared__ __align__(16) float shCN[KCODES];    // -||c||^2 bias source (4KB)
    const int tid  = threadIdx.x;
    const int lane = tid & 31;
    const int wave = tid >> 5;
    const int gtile = blockIdx.x * 8 + wave;

    // Stage this codebook's norms into LDS (256 x float4 = 1024 floats).
    ((float4*)shCN)[tid] = ((const float4*)(cnorm + q * KCODES))[tid];

    // Load this wave's A strip once; reused across the whole K loop.
    BF16x16 A[4];
    const v4u* aptr = (const v4u*)(apack + (size_t)gtile * TILE_USH);
#pragma unroll
    for (int cc = 0; cc < 4; ++cc) {
        A[cc].u[0] = aptr[(cc * 32 + lane) * 2 + 0];
        A[cc].u[1] = aptr[(cc * 32 + lane) * 2 + 1];
    }

    const unsigned short* bbase = bpack + (size_t)q * (64 * TILE_USH);
    // Prologue: tile 0 -> buffer 0.
    shB[0][tid] = ((const v4u*)bbase)[tid];

    float bestv[8];
    int   besti[8];
#pragma unroll
    for (int j = 0; j < 8; ++j) { bestv[j] = -3.4e38f; besti[j] = 0; }

    for (int kb = 0; kb < 64; ++kb) {
        const int buf = kb & 1;
        __syncthreads();                            // tile kb visible in shB[buf]

        // Issue next tile's global load early; consumed after the WMMA chain.
        v4u nxt;
        const bool hasNext = (kb + 1 < 64);
        if (hasNext)
            nxt = ((const v4u*)(bbase + (size_t)(kb + 1) * TILE_USH))[tid];
        if (kb + 2 < 64)
            __builtin_prefetch(bbase + (size_t)(kb + 2) * TILE_USH, 0, 1);

        float ci = -0.5f * shCN[kb * 16 + (lane & 15)];

        // Preload all 4 B chunks -> ds clause, then 4 back-to-back WMMAs.
        BF16x16 Bt[4];
#pragma unroll
        for (int cc = 0; cc < 4; ++cc) {
            Bt[cc].u[0] = shB[buf][(cc * 32 + lane) * 2 + 0];
            Bt[cc].u[1] = shB[buf][(cc * 32 + lane) * 2 + 1];
        }

        v8f acc;
#pragma unroll
        for (int j = 0; j < 8; ++j) acc[j] = ci;
#pragma unroll
        for (int cc = 0; cc < 4; ++cc) {
            acc = __builtin_amdgcn_wmma_f32_16x16x32_bf16(
                false, A[cc].v, false, Bt[cc].v, (short)0, acc, false, false);
        }

        int code = kb * 16 + (lane & 15);
#pragma unroll
        for (int j = 0; j < 8; ++j) {
            if (acc[j] > bestv[j]) { bestv[j] = acc[j]; besti[j] = code; }
        }

        if (hasNext) shB[buf ^ 1][tid] = nxt;       // loadcnt wait lands here
    }

    // Cross-lane argmax within each 16-lane half (C/D layout: vgpr j ->
    // M=j for lanes 0-15, M=j+8 for lanes 16-31). Tie -> smallest index.
#pragma unroll
    for (int j = 0; j < 8; ++j) {
        float bv = bestv[j];
        int   bi = besti[j];
#pragma unroll
        for (int off = 8; off >= 1; off >>= 1) {
            float ov = __shfl_xor(bv, off, 32);
            int   oi = __shfl_xor(bi, off, 32);
            if (ov > bv || (ov == bv && oi < bi)) { bv = ov; bi = oi; }
        }
        if ((lane & 15) == 0) {
            int row = j + ((lane >> 4) << 3);
            idx_all[(size_t)(gtile * 16 + row) * QSTAGES + q] = bi;
        }
    }
}

// Wave-per-row update: each lane owns one float4 of the 128-dim row.
// Coalesced b128 traffic; bf16 repack as one aligned b64 store per lane.
__global__ __launch_bounds__(256) void rvq_update(
        const float* __restrict__ cb, const int* __restrict__ idx_all,
        float* __restrict__ res, float* __restrict__ qout,
        unsigned short* __restrict__ apack,
        float* __restrict__ lossSum, int* __restrict__ counts, int q) {
    const int lane = threadIdx.x & 31;
    const int wave = threadIdx.x >> 5;
    const int m = blockIdx.x * 8 + wave;

    const int idx = idx_all[(size_t)m * QSTAGES + q];
    const float4* crow = (const float4*)(cb + ((size_t)q * KCODES + idx) * CDIM);
    float4* rrow = (float4*)(res  + (size_t)m * CDIM);
    float4* qrow = (float4*)(qout + (size_t)m * CDIM);

    float4 r  = rrow[lane];
    float4 xd = crow[lane];
    float4 qa = qrow[lane];
    float4 d;
    d.x = r.x - xd.x; d.y = r.y - xd.y; d.z = r.z - xd.z; d.w = r.w - xd.w;
    float ls = d.x * d.x + d.y * d.y + d.z * d.z + d.w * d.w;
    qa.x += xd.x; qa.y += xd.y; qa.z += xd.z; qa.w += xd.w;
    qrow[lane] = qa;
    rrow[lane] = d;

    // Repack new residual to bf16 A layout: c = lane*4 .. lane*4+3 are
    // 4 consecutive elems in the same packed lane -> one aligned b64 store.
    uint2 uu;
    uu.x = (unsigned int)f2bf(d.x) | ((unsigned int)f2bf(d.y) << 16);
    uu.y = (unsigned int)f2bf(d.z) | ((unsigned int)f2bf(d.w) << 16);
    *(uint2*)(apack + apack_off(m, lane * 4)) = uu;

    // Wave-reduce the squared-error sum; one atomic per row.
#pragma unroll
    for (int off = 16; off >= 1; off >>= 1) ls += __shfl_xor(ls, off, 32);
    if (lane == 0) {
        atomicAdd(&lossSum[q], ls * (1.0f / ((float)MROWS * (float)CDIM)));
        atomicAdd(&counts[q * KCODES + idx], 1);
    }
}

__global__ __launch_bounds__(1024) void rvq_perplexity(const int* __restrict__ counts,
                                                       float* __restrict__ perp) {
    __shared__ float sh[1024];
    int q = blockIdx.x, k = threadIdx.x;
    float p = (float)counts[q * KCODES + k] * (1.0f / (float)MROWS);
    sh[k] = p * logf(p + EPSV);
    __syncthreads();
    for (int s = 512; s > 0; s >>= 1) {
        if (k < s) sh[k] += sh[k + s];
        __syncthreads();
    }
    if (k == 0) perp[q] = expf(-sh[0]);
}

__global__ void rvq_finalize(const float* __restrict__ lossSum,
                             const float* __restrict__ perp,
                             float* __restrict__ out_scalars) {
    float l = 0.f, p = 0.f;
    for (int q = 0; q < QSTAGES; ++q) { l += lossSum[q]; p += perp[q]; }
    out_scalars[0] = l * (1.0f / QSTAGES);
    out_scalars[1] = p * (1.0f / QSTAGES);
}

__global__ __launch_bounds__(256) void rvq_write_quant(const float* __restrict__ qout,
                                                       float* __restrict__ out) {
    size_t e = (size_t)blockIdx.x * 256 + threadIdx.x;   // over B*C*J*T
    int t = (int)(e % TDIM); size_t r = e / TDIM;
    int j = (int)(r % JDIM); r /= JDIM;
    int c = (int)(r % CDIM); int b = (int)(r / CDIM);
    int m = (b * JDIM + j) * TDIM + t;
    out[e] = qout[(size_t)m * CDIM + c];
}

__global__ __launch_bounds__(256) void rvq_write_idx(const int* __restrict__ idx_all,
                                                     float* __restrict__ out) {
    int i = blockIdx.x * 256 + threadIdx.x;
    if (i < (int)(MROWS * QSTAGES)) out[i] = (float)idx_all[i];
}

// ---------------- workspace layout (bytes) ----------------
#define OFF_RES    ((size_t)0)
#define SZ_RES     ((size_t)MROWS * CDIM * 4)                 // 50,331,648
#define OFF_QOUT   (OFF_RES + SZ_RES)
#define SZ_QOUT    SZ_RES
#define OFF_APACK  (OFF_QOUT + SZ_QOUT)
#define SZ_APACK   ((size_t)MROWS * CDIM * 2)
#define OFF_BPACK  (OFF_APACK + SZ_APACK)
#define SZ_BPACK   ((size_t)QSTAGES * KCODES * CDIM * 2)
#define OFF_CNORM  (OFF_BPACK + SZ_BPACK)
#define SZ_CNORM   ((size_t)QSTAGES * KCODES * 4)
#define OFF_IDX    (OFF_CNORM + SZ_CNORM)
#define SZ_IDX     ((size_t)MROWS * QSTAGES * 4)
#define OFF_COUNTS (OFF_IDX + SZ_IDX)
#define SZ_COUNTS  ((size_t)QSTAGES * KCODES * 4)
#define OFF_LOSS   (OFF_COUNTS + SZ_COUNTS)
#define SZ_LOSS    ((size_t)QSTAGES * 4)
#define OFF_PERP   (OFF_LOSS + SZ_LOSS)
#define SZ_PERP    ((size_t)QSTAGES * 4)

extern "C" void kernel_launch(void* const* d_in, const int* in_sizes, int n_in,
                              void* d_out, int out_size, void* d_ws, size_t ws_size,
                              hipStream_t stream) {
    const float* x  = (const float*)d_in[0];   // (B,C,J,T) fp32
    const float* cb = (const float*)d_in[1];   // (Q,K,C) fp32
    float* out = (float*)d_out;
    char*  ws  = (char*)d_ws;

    float*          res    = (float*)(ws + OFF_RES);
    float*          qout   = (float*)(ws + OFF_QOUT);
    unsigned short* apack  = (unsigned short*)(ws + OFF_APACK);
    unsigned short* bpack  = (unsigned short*)(ws + OFF_BPACK);
    float*          cnorm  = (float*)(ws + OFF_CNORM);
    int*            idxall = (int*)(ws + OFF_IDX);
    int*            counts = (int*)(ws + OFF_COUNTS);
    float*          lossS  = (float*)(ws + OFF_LOSS);
    float*          perp   = (float*)(ws + OFF_PERP);

    // fresh accumulators every call (graph-capture safe)
    hipMemsetAsync(qout, 0, SZ_QOUT, stream);
    hipMemsetAsync(counts, 0, SZ_COUNTS + SZ_LOSS + SZ_PERP, stream);

    rvq_pack_x<<<(MROWS * CDIM) / 256, 256, 0, stream>>>(x, res, apack);
    rvq_pack_cb<<<(QSTAGES * KCODES + 255) / 256, 256, 0, stream>>>(cb, bpack, cnorm);

    for (int q = 0; q < QSTAGES; ++q) {
        rvq_dist_argmin<<<MTILES / 8, 256, 0, stream>>>(apack, bpack, cnorm, idxall, q);
        rvq_update<<<MROWS / 8, 256, 0, stream>>>(cb, idxall, res, qout, apack,
                                                  lossS, counts, q);
    }

    rvq_perplexity<<<QSTAGES, 1024, 0, stream>>>(counts, perp);
    rvq_finalize<<<1, 1, 0, stream>>>(lossS, perp, out + NQOUT + NIDX);
    rvq_write_quant<<<(int)(NQOUT / 256), 256, 0, stream>>>(qout, out);
    rvq_write_idx<<<(int)((NIDX + 255) / 256), 256, 0, stream>>>(idxall, out + NQOUT);
}
